// PEARLPositionalEncoder_16372415332568
// MI455X (gfx1250) — compile-verified, matrix operand order checked by reading, not run on hardware
//
#include <hip/hip_runtime.h>
#include <stdint.h>

// ---------------------------------------------------------------------------
// Problem constants (match reference)
// ---------------------------------------------------------------------------
#define Bn   8
#define Nn   1024
#define Mn   128
#define KP   16      // number of Laplacian powers
#define Cc   16      // MLP_OUT channels
#define DPE  64      // phi output dim
#define En   131072  // edges
#define NM   (Nn * Mn)

typedef __attribute__((ext_vector_type(16))) __bf16 v16bf;
typedef __attribute__((ext_vector_type(8)))  __bf16 v8bf;
typedef __attribute__((ext_vector_type(8)))  float  v8f;

// ---------------------------------------------------------------------------
// Split fp32 -> bf16 hi/lo (bf16x3 emulation operands). Optionally copy the
// f32 source into feat slot k=0 (with per-batch stride remap).
// ---------------------------------------------------------------------------
__global__ void pearl_split_kernel(const float* __restrict__ src,
                                   __bf16* __restrict__ hi,
                                   __bf16* __restrict__ lo,
                                   float* __restrict__ featCopy,
                                   long long n) {
  long long i = (long long)blockIdx.x * blockDim.x + threadIdx.x;
  if (i >= n) return;
  float v  = src[i];
  __bf16 h = (__bf16)v;
  hi[i] = h;
  lo[i] = (__bf16)(v - (float)h);
  if (featCopy) {
    // src is [B][N][M]; feat is [B][KP][N][M], slot k=0
    long long b = i / (long long)NM;
    featCopy[b * (long long)(KP - 1) * NM + i] = v;
  }
}

// ---------------------------------------------------------------------------
// Zero per-launch accumulators (agg + stats). Deterministic re-init each call.
// ---------------------------------------------------------------------------
__global__ void pearl_zero_kernel(float* __restrict__ agg, float* __restrict__ stats) {
  int i = blockIdx.x * blockDim.x + threadIdx.x;
  if (i < Bn * Nn * Cc) agg[i] = 0.f;
  if (i < Bn * Cc * 2)  stats[i] = 0.f;
}

// ---------------------------------------------------------------------------
// One step of the power chain: Y[b] = S[b] @ X[b]  (1024x1024 @ 1024x128)
// bf16x3 emulated fp32: Y = Shi@Xhi + Shi@Xlo + Slo@Xhi (f32 accumulate).
// Block = 256 threads (8 waves). Block tile: 128 rows x 64 cols.
// Wave w -> rows [w*16, w*16+16), all 64 cols (4 WMMA tiles).
//
// LDS layout trick: the 32 K-values of each row are stored permuted
// [0..7, 16..23, 8..15, 24..31] so that a lane's WMMA fragment (ISA 7.12.2
// 16-bit A layout: lanes 0-15 take K {0..7,16..23}, lanes 16-31 take
// K {8..15,24..31}) is ONE contiguous, 32B-aligned v16bf -> ds_load_b128
// pairs instead of 16 scalar ds_load_u16. X is staged transposed
// (sXT[col][perm(k)]) so B fragments are contiguous too.
// Row stride 48 bf16 = 96B = 3*32B keeps 32B alignment and spreads banks.
// ---------------------------------------------------------------------------
__global__ __launch_bounds__(256)
void pearl_gemm_kernel(const __bf16* __restrict__ Shi,
                       const __bf16* __restrict__ Slo,
                       const __bf16* __restrict__ Xhi,
                       const __bf16* __restrict__ Xlo,
                       __bf16* __restrict__ Yhi,
                       __bf16* __restrict__ Ylo,
                       float* __restrict__ feat,
                       int kstep) {
  const int b    = blockIdx.z;
  const int row0 = blockIdx.x * 128;
  const int col0 = blockIdx.y * 64;

  const __bf16* Sh = Shi + (size_t)b * Nn * Nn;
  const __bf16* Sl = Slo + (size_t)b * Nn * Nn;
  const __bf16* Xh = Xhi + (size_t)b * NM;
  const __bf16* Xl = Xlo + (size_t)b * NM;
  float* featPtr   = feat + ((size_t)b * KP + kstep) * NM;
  __bf16* Yh       = Yhi + (size_t)b * NM;
  __bf16* Yl       = Ylo + (size_t)b * NM;

  __shared__ __attribute__((aligned(32))) __bf16 sAh[128][48];  // K-permuted rows
  __shared__ __attribute__((aligned(32))) __bf16 sAl[128][48];
  __shared__ __attribute__((aligned(32))) __bf16 sXTh[64][48];  // transposed, K-permuted
  __shared__ __attribute__((aligned(32))) __bf16 sXTl[64][48];

  const int tid   = threadIdx.x;
  const int wave  = tid >> 5;       // 0..7
  const int lane  = tid & 31;
  const int l16   = lane & 15;
  const int khalf = lane >> 4;      // fragment half per ISA A/B lane layout

  // staging roles
  const int ar  = tid >> 1;          // A: row 0..127
  const int ac0 = (tid & 1) << 4;    // A: k-chunk half, 0 or 16
  const int alo = (ac0 == 0) ? 0 : 8;  // permuted low-run destination
  const int xkr = tid >> 3;          // X: k row 0..31
  const int xc0 = (tid & 7) << 3;    // X: col chunk 0..56
  const int lutp[4] = {0, 16, 8, 24};
  const int xp  = lutp[xkr >> 3] + (xkr & 7);  // permuted k position

  v8f acc[4];
#pragma unroll
  for (int t = 0; t < 4; ++t) acc[t] = (v8f){0.f, 0.f, 0.f, 0.f, 0.f, 0.f, 0.f, 0.f};

  for (int kc = 0; kc < Nn; kc += 32) {
    // --- prefetch next S panel into caches (global_prefetch_b8) ---
    if (kc + 32 < Nn) {
      __builtin_prefetch(Sh + (size_t)(row0 + ar) * Nn + (kc + 32) + ac0, 0, 0);
      __builtin_prefetch(Sl + (size_t)(row0 + ar) * Nn + (kc + 32) + ac0, 0, 0);
    }

    // --- stage A (S tile 128x32, hi+lo), vector loads + permuted vector stores
    {
      size_t g = (size_t)(row0 + ar) * Nn + kc + ac0;
      v16bf vh = *(const v16bf*)(Sh + g);
      v16bf vl = *(const v16bf*)(Sl + g);
      v8bf vh0 = __builtin_shufflevector(vh, vh, 0, 1, 2, 3, 4, 5, 6, 7);
      v8bf vh1 = __builtin_shufflevector(vh, vh, 8, 9, 10, 11, 12, 13, 14, 15);
      v8bf vl0 = __builtin_shufflevector(vl, vl, 0, 1, 2, 3, 4, 5, 6, 7);
      v8bf vl1 = __builtin_shufflevector(vl, vl, 8, 9, 10, 11, 12, 13, 14, 15);
      *(v8bf*)&sAh[ar][alo]      = vh0;
      *(v8bf*)&sAh[ar][alo + 16] = vh1;
      *(v8bf*)&sAl[ar][alo]      = vl0;
      *(v8bf*)&sAl[ar][alo + 16] = vl1;
    }

    // --- stage X transposed (32x64 tile -> sXT[col][perm(k)], hi+lo) ---
    {
      size_t g = (size_t)(kc + xkr) * Mn + col0 + xc0;
      v8bf xh = *(const v8bf*)(Xh + g);
      v8bf xl = *(const v8bf*)(Xl + g);
#pragma unroll
      for (int j = 0; j < 8; ++j) {
        sXTh[xc0 + j][xp] = xh[j];
        sXTl[xc0 + j][xp] = xl[j];
      }
    }
    __syncthreads();

    // --- A fragments: single contiguous 32B per lane ---
    const int arow = wave * 16 + l16;
    v16bf ah = *(const v16bf*)&sAh[arow][khalf * 16];
    v16bf al = *(const v16bf*)&sAl[arow][khalf * 16];

    // --- 4 column tiles x 3 WMMAs (bf16x3) ---
#pragma unroll
    for (int t = 0; t < 4; ++t) {
      const int bcol = t * 16 + l16;
      v16bf bh = *(const v16bf*)&sXTh[bcol][khalf * 16];
      v16bf bl = *(const v16bf*)&sXTl[bcol][khalf * 16];
      acc[t] = __builtin_amdgcn_wmma_f32_16x16x32_bf16(false, ah, false, bh,
                                                       (short)0, acc[t], false, false);
      acc[t] = __builtin_amdgcn_wmma_f32_16x16x32_bf16(false, ah, false, bl,
                                                       (short)0, acc[t], false, false);
      acc[t] = __builtin_amdgcn_wmma_f32_16x16x32_bf16(false, al, false, bh,
                                                       (short)0, acc[t], false, false);
    }
    __syncthreads();
  }

  // --- epilogue: C/D layout (ISA 7.12.2): lanes 0-15 M=r, lanes 16-31 M=8+r ---
  const int mbase = (lane < 16) ? 0 : 8;
#pragma unroll
  for (int t = 0; t < 4; ++t) {
#pragma unroll
    for (int r = 0; r < 8; ++r) {
      int row = row0 + wave * 16 + mbase + r;
      int col = col0 + t * 16 + l16;
      float v = acc[t][r];
      size_t off = (size_t)row * Mn + col;
      featPtr[off] = v;
      __bf16 h = (__bf16)v;
      Yh[off] = h;
      Yl[off] = (__bf16)(v - (float)h);
    }
  }
}

// ---------------------------------------------------------------------------
// Per-(b,c) sum and sum-of-squares of h = feat . lin_w + lin_b over (N,M).
// Grid: (64 chunks, B). 256 threads, 8 points each. Wave shuffle reduction,
// then global atomics.
// ---------------------------------------------------------------------------
__global__ __launch_bounds__(256)
void pearl_stats_kernel(const float* __restrict__ feat,
                        const float* __restrict__ lin_w,
                        const float* __restrict__ lin_b,
                        float* __restrict__ stats) {
  const int b   = blockIdx.y;
  const int tid = threadIdx.x;
  __shared__ float w[KP * Cc];
  __shared__ float bsh[Cc];
  if (tid < KP * Cc) w[tid] = lin_w[tid];
  if (tid < Cc)      bsh[tid] = lin_b[tid];
  __syncthreads();

  const float* fb = feat + (size_t)b * KP * NM;
  float sum[Cc], sq[Cc];
#pragma unroll
  for (int c = 0; c < Cc; ++c) { sum[c] = 0.f; sq[c] = 0.f; }

  const int base = blockIdx.x * 256 * 8;
  for (int p = 0; p < 8; ++p) {
    int idx = base + p * 256 + tid;
    float f[KP];
#pragma unroll
    for (int k = 0; k < KP; ++k) f[k] = fb[(size_t)k * NM + idx];
#pragma unroll
    for (int c = 0; c < Cc; ++c) {
      float h = bsh[c];
#pragma unroll
      for (int k = 0; k < KP; ++k) h += f[k] * w[k * Cc + c];
      sum[c] += h;
      sq[c]  += h * h;
    }
  }

  // wave32 reduction
#pragma unroll
  for (int c = 0; c < Cc; ++c) {
    for (int off = 16; off > 0; off >>= 1) {
      sum[c] += __shfl_xor(sum[c], off, 32);
      sq[c]  += __shfl_xor(sq[c],  off, 32);
    }
  }
  if ((tid & 31) == 0) {
#pragma unroll
    for (int c = 0; c < Cc; ++c) {
      atomicAdd(&stats[b * 2 * Cc + c],      sum[c]);
      atomicAdd(&stats[b * 2 * Cc + Cc + c], sq[c]);
    }
  }
}

// ---------------------------------------------------------------------------
// Normalize (per-graph BN, training mode) + ReLU + mean over M.
// Grid: (N, B). 128 threads = one per m. x[b*N+n][c] written.
// ---------------------------------------------------------------------------
__global__ __launch_bounds__(128)
void pearl_norm_pool_kernel(const float* __restrict__ feat,
                            const float* __restrict__ lin_w,
                            const float* __restrict__ lin_b,
                            const float* __restrict__ gamma,
                            const float* __restrict__ beta,
                            const float* __restrict__ stats,
                            float* __restrict__ xbuf) {
  const int b = blockIdx.y, n = blockIdx.x, tid = threadIdx.x;
  __shared__ float w[KP * Cc];
  __shared__ float scale[Cc], shift[Cc], bsh[Cc];
  __shared__ float red[4 * Cc];

  w[tid] = lin_w[tid];
  w[tid + 128] = lin_w[tid + 128];
  if (tid < Cc) {
    const float inv = 1.f / (float)NM;
    float mean = stats[b * 2 * Cc + tid] * inv;
    float var  = stats[b * 2 * Cc + Cc + tid] * inv - mean * mean;
    float rs   = rsqrtf(var + 1e-5f) * gamma[tid];
    scale[tid] = rs;
    shift[tid] = beta[tid] - mean * rs;
    bsh[tid]   = lin_b[tid];
  }
  __syncthreads();

  const float* fb = feat + (size_t)b * KP * NM + (size_t)n * Mn + tid;
  float f[KP];
#pragma unroll
  for (int k = 0; k < KP; ++k) f[k] = fb[(size_t)k * NM];

  float v[Cc];
#pragma unroll
  for (int c = 0; c < Cc; ++c) {
    float h = bsh[c];
#pragma unroll
    for (int k = 0; k < KP; ++k) h += f[k] * w[k * Cc + c];
    h = h * scale[c] + shift[c];
    v[c] = fmaxf(h, 0.f) * (1.f / (float)Mn);
  }

  const int wv = tid >> 5, ln = tid & 31;
#pragma unroll
  for (int c = 0; c < Cc; ++c)
    for (int off = 16; off > 0; off >>= 1) v[c] += __shfl_xor(v[c], off, 32);
  if (ln == 0) {
#pragma unroll
    for (int c = 0; c < Cc; ++c) red[wv * Cc + c] = v[c];
  }
  __syncthreads();
  if (tid < Cc) {
    float s = red[tid] + red[Cc + tid] + red[2 * Cc + tid] + red[3 * Cc + tid];
    xbuf[((size_t)b * Nn + n) * Cc + tid] = s;
  }
}

// ---------------------------------------------------------------------------
// GIN segment-sum: agg[dst] += x[src] over 131072 edges (float atomics).
// ---------------------------------------------------------------------------
__global__ void pearl_edges_kernel(const long long* __restrict__ ei,
                                   const float* __restrict__ xbuf,
                                   float* __restrict__ agg) {
  int e = blockIdx.x * blockDim.x + threadIdx.x;
  if (e >= En) return;
  int s = (int)ei[e];
  int d = (int)ei[En + e];
#pragma unroll
  for (int c = 0; c < Cc; ++c)
    atomicAdd(&agg[(size_t)d * Cc + c], xbuf[(size_t)s * Cc + c]);
}

// ---------------------------------------------------------------------------
// phi MLP: out = relu((x+agg) @ w1 + b1) @ w2 + b2. One block per row.
// ---------------------------------------------------------------------------
__global__ __launch_bounds__(64)
void pearl_phi_kernel(const float* __restrict__ xbuf,
                      const float* __restrict__ agg,
                      const float* __restrict__ w1,
                      const float* __restrict__ b1,
                      const float* __restrict__ w2,
                      const float* __restrict__ b2,
                      float* __restrict__ out) {
  const int row = blockIdx.x, j = threadIdx.x;
  __shared__ float g[Cc];
  __shared__ float hid[DPE];
  if (j < Cc) g[j] = xbuf[(size_t)row * Cc + j] + agg[(size_t)row * Cc + j];
  __syncthreads();
  float h = b1[j];
#pragma unroll
  for (int c = 0; c < Cc; ++c) h += g[c] * w1[c * DPE + j];
  hid[j] = fmaxf(h, 0.f);
  __syncthreads();
  float o = b2[j];
  for (int i = 0; i < DPE; ++i) o += hid[i] * w2[i * DPE + j];
  out[(size_t)row * DPE + j] = o;
}

// ---------------------------------------------------------------------------
// Launcher
// ---------------------------------------------------------------------------
extern "C" void kernel_launch(void* const* d_in, const int* in_sizes, int n_in,
                              void* d_out, int out_size, void* d_ws, size_t ws_size,
                              hipStream_t stream) {
  const float*     Lap  = (const float*)d_in[0];
  const float*     W    = (const float*)d_in[1];
  const float*     linw = (const float*)d_in[2];
  const float*     linb = (const float*)d_in[3];
  const float*     gam  = (const float*)d_in[4];
  const float*     bet  = (const float*)d_in[5];
  const float*     pw1  = (const float*)d_in[6];
  const float*     pb1  = (const float*)d_in[7];
  const float*     pw2  = (const float*)d_in[8];
  const float*     pb2  = (const float*)d_in[9];
  const long long* ei   = (const long long*)d_in[10];
  float*           out  = (float*)d_out;

  // ---- workspace carve-up (~106 MB) ----
  __bf16* sHi = (__bf16*)d_ws;                          // 8*1024*1024 bf16
  __bf16* sLo = sHi + (size_t)Bn * Nn * Nn;
  __bf16* xh0 = sLo + (size_t)Bn * Nn * Nn;             // ping-pong X hi/lo
  __bf16* xl0 = xh0 + (size_t)Bn * NM;
  __bf16* xh1 = xl0 + (size_t)Bn * NM;
  __bf16* xl1 = xh1 + (size_t)Bn * NM;
  float*  feat  = (float*)(xl1 + (size_t)Bn * NM);      // [B][KP][N][M] f32
  float*  stats = feat + (size_t)Bn * KP * NM;          // [B][2][Cc]
  float*  xbuf  = stats + Bn * 2 * Cc;                  // [B*N][Cc]
  float*  agg   = xbuf + (size_t)Bn * Nn * Cc;          // [B*N][Cc]

  // 1) zero accumulators
  pearl_zero_kernel<<<(Bn * Nn * Cc + 255) / 256, 256, 0, stream>>>(agg, stats);

  // 2) split S and W into bf16 hi/lo; copy W into feat[:,0]
  {
    long long nS = (long long)Bn * Nn * Nn;
    pearl_split_kernel<<<(unsigned)((nS + 255) / 256), 256, 0, stream>>>(
        Lap, sHi, sLo, nullptr, nS);
    long long nW = (long long)Bn * NM;
    pearl_split_kernel<<<(unsigned)((nW + 255) / 256), 256, 0, stream>>>(
        W, xh0, xl0, feat, nW);
  }

  // 3) sequential power chain, 15 WMMA GEMM steps (ping-pong X buffers)
  __bf16* xh[2] = {xh0, xh1};
  __bf16* xl[2] = {xl0, xl1};
  for (int k = 1; k < KP; ++k) {
    int pi = (k - 1) & 1, po = k & 1;
    pearl_gemm_kernel<<<dim3(Nn / 128, Mn / 64, Bn), 256, 0, stream>>>(
        sHi, sLo, xh[pi], xl[pi], xh[po], xl[po], feat, k);
  }

  // 4) BN statistics per (b, c)
  pearl_stats_kernel<<<dim3(64, Bn), 256, 0, stream>>>(feat, linw, linb, stats);

  // 5) normalize + ReLU + mean over M -> x
  pearl_norm_pool_kernel<<<dim3(Nn, Bn), 128, 0, stream>>>(
      feat, linw, linb, gam, bet, stats, xbuf);

  // 6) edge segment sum (GIN aggregation)
  pearl_edges_kernel<<<(En + 255) / 256, 256, 0, stream>>>(ei, xbuf, agg);

  // 7) phi MLP -> out [8192, 64]
  pearl_phi_kernel<<<Bn * Nn, 64, 0, stream>>>(xbuf, agg, pw1, pb1, pw2, pb2, out);
}
